// ContTimeLSTM_66623532695635
// MI455X (gfx1250) — matrix operand order repaired and checked by reading, not compiled, not run on hardware
//
#include <hip/hip_runtime.h>
#include <hip/hip_bf16.h>
#include <math.h>

typedef __attribute__((ext_vector_type(16))) _Float16 v16h;
typedef __attribute__((ext_vector_type(8)))  _Float16 v8h;
typedef __attribute__((ext_vector_type(8)))  float    v8f;
typedef int v4i_ __attribute__((vector_size(16)));

#define B_    128
#define L_    512
#define I_    256
#define D_    512
#define K_    768
#define NC_   7
#define WPAD  776      // 768 + 8 halves pad -> LDS bank spread
#define NWG   64       // 32 j-slices x 2 m-slices
#define TPB   128      // 4 waves of 32

#if __has_builtin(__builtin_amdgcn_global_load_async_to_lds_b128)
#define HAVE_ASYNC_LDS 1
#endif

__device__ __forceinline__ float sigmoidf_(float x) { return 1.f / (1.f + __expf(-x)); }
__device__ __forceinline__ float softplusf_(float x) { return (x > 20.f) ? x : log1pf(__expf(x)); }

__device__ __forceinline__ void grid_sync(unsigned* cnt, unsigned* gen, unsigned nwg) {
    __threadfence();                 // release our stores to device scope
    __syncthreads();
    if (threadIdx.x == 0) {
        unsigned g = __hip_atomic_load(gen, __ATOMIC_RELAXED, __HIP_MEMORY_SCOPE_AGENT);
        unsigned prev = __hip_atomic_fetch_add(cnt, 1u, __ATOMIC_ACQ_REL, __HIP_MEMORY_SCOPE_AGENT);
        if (prev == nwg - 1u) {
            __hip_atomic_store(cnt, 0u, __ATOMIC_RELAXED, __HIP_MEMORY_SCOPE_AGENT);
            __hip_atomic_fetch_add(gen, 1u, __ATOMIC_ACQ_REL, __HIP_MEMORY_SCOPE_AGENT);
        } else {
            while (__hip_atomic_load(gen, __ATOMIC_ACQUIRE, __HIP_MEMORY_SCOPE_AGENT) == g) {
                __builtin_amdgcn_s_sleep(2);
            }
        }
    }
    __syncthreads();
    __threadfence();                 // acquire other blocks' stores
}

// Convert masked x to f16 once: x_f16[b,t,i] = (t < seq[b]) ? x : 0
__global__ __launch_bounds__(256) void prep_x(const float* __restrict__ x,
                                              const int* __restrict__ seq,
                                              _Float16* __restrict__ xh) {
    const long long nel = (long long)B_ * L_ * I_;
    for (long long e = (long long)blockIdx.x * 256 + threadIdx.x; e < nel;
         e += (long long)gridDim.x * 256) {
        int b = (int)(e / ((long long)L_ * I_));
        int t = (int)((e / I_) % L_);
        float v = (t < seq[b]) ? x[e] : 0.f;
        xh[e] = (_Float16)v;
    }
}

// Convert W to f16 once (staged into LDS by the persistent kernel)
__global__ __launch_bounds__(256) void prep_w(const float* __restrict__ W,
                                              _Float16* __restrict__ wh) {
    const long long nel = (long long)NC_ * D_ * K_;
    for (long long e = (long long)blockIdx.x * 256 + threadIdx.x; e < nel;
         e += (long long)gridDim.x * 256) {
        wh[e] = (_Float16)W[e];
    }
}

// BOS step from zero state: c = 0, h = 0 -> proj = bos @ Wx^T + bias
__global__ __launch_bounds__(256) void init_state(const float* __restrict__ bos,
                                                  const float* __restrict__ W,
                                                  const float* __restrict__ bias,
                                                  float* st_o, float* st_cs,
                                                  float* st_ce, float* st_d) {
    int j = blockIdx.x * 256 + threadIdx.x;
    if (j >= D_) return;
    float p[7];
    for (int c = 0; c < 7; ++c) {
        const float* wr = W + (size_t)(c * D_ + j) * K_;
        float s = bias[c * D_ + j];
        for (int i = 0; i < I_; ++i) s += bos[i] * wr[i];
        p[c] = s;
    }
    float i_g  = sigmoidf_(p[0]);
    float ie_g = sigmoidf_(p[2]);
    float o_n  = sigmoidf_(p[4]);
    float z    = tanhf(p[5]);
    float d_n  = softplusf_(p[6]);
    float cs_n = i_g * z;    // f_g * 0 + i_g * z
    float ce_n = ie_g * z;   // fe_g * 0 + ie_g * z
    for (int b = 0; b < B_; ++b) {
        st_o[b * D_ + j] = o_n;   st_cs[b * D_ + j] = cs_n;
        st_ce[b * D_ + j] = ce_n; st_d[b * D_ + j] = d_n;
    }
}

__global__ __launch_bounds__(TPB) void ct_lstm_persistent(
    const _Float16* __restrict__ xh, const _Float16* __restrict__ wh,
    _Float16* __restrict__ hh0, _Float16* __restrict__ hh1,
    const float* st_o, const float* st_cs, const float* st_ce, const float* st_d,
    const float* __restrict__ dt, const int* __restrict__ seq,
    const float* __restrict__ bias,
    float* __restrict__ out, unsigned* bar) {
    extern __shared__ _Float16 smemh[];   // [7*16][WPAD] f16 weight slice
    const int tid    = threadIdx.x;
    const int wg     = blockIdx.x;
    const int jb     = wg & 31;           // j slice
    const int mb     = wg >> 5;           // batch-row slice
    const int j0     = jb * 16;
    const int m0     = mb * 64 + (tid >> 5) * 16;   // wave's 16-row M tile
    const int lane   = tid & 31;
    const int hlf    = lane >> 4;
    const int n      = lane & 15;
    const int mrow   = m0 + n;
    unsigned* cnt = bar;
    unsigned* gen = bar + 32;

    // ---- Stage this block's W slice (7 chunks x 16 cols x 768 K, f16) into LDS.
    // 16-byte granules: idx -> rc = row (c*16+r), k = K offset.
    for (int idx = tid; idx < NC_ * 16 * (K_ / 8); idx += TPB) {
        int rc = idx / (K_ / 8);
        int k  = (idx % (K_ / 8)) * 8;
        int col = (rc >> 4) * D_ + j0 + (rc & 15);
        const _Float16* src = wh + (size_t)col * K_ + k;
        _Float16* dst = smemh + (size_t)rc * WPAD + k;
#if HAVE_ASYNC_LDS
        __builtin_amdgcn_global_load_async_to_lds_b128((v4i_*)src, (v4i_*)dst, 0, 0);
#else
        *(v8h*)dst = *(const v8h*)src;
#endif
    }
#if HAVE_ASYNC_LDS
    asm volatile("s_wait_asynccnt 0x0" ::: "memory");
#endif
    float biasv[7];
#pragma unroll
    for (int c = 0; c < 7; ++c) biasv[c] = bias[c * D_ + j0 + n];
    __syncthreads();

    float* outF = out + (size_t)B_ * L_ * D_;   // final states region (B, 4D)

    // ---- Prologue: per-thread register states for 8 (b, j) elements; emit h_0.
    int   bi[8], ei[8], sq[8], la[8];
    float ro[8], rcs[8], rce[8], rd[8], rc[8];
#pragma unroll
    for (int r = 0; r < 8; ++r) {
        int b_idx = m0 + hlf * 8 + r;
        bi[r] = b_idx;
        ei[r] = b_idx * D_ + j0 + n;
        int s = seq[b_idx];
        sq[r] = s;
        la[r] = (s > 1) ? (s - 1) : 0;
        ro[r]  = st_o[ei[r]];
        rcs[r] = st_cs[ei[r]];
        rce[r] = st_ce[ei[r]];
        rd[r]  = st_d[ei[r]];
        float dt0 = (0 < s) ? dt[(size_t)b_idx * L_] : 0.f;
        float c0  = rce[r] + (rcs[r] - rce[r]) * __expf(-rd[r] * dt0);
        rc[r] = c0;
        float h0 = ro[r] * tanhf(c0);
        hh0[ei[r]] = (_Float16)h0;
        out[(size_t)b_idx * L_ * D_ + (j0 + n)] = h0;
    }
    grid_sync(cnt, gen, NWG);

    for (int t = 0; t < L_; ++t) {
        const _Float16* hcur = (t & 1) ? hh1 : hh0;
        _Float16*       hnxt = (t & 1) ? hh0 : hh1;

        // ---- GEMM: proj = [x_t | h_t] @ W_slice^T (M=16 rows/wave, 7 N-tiles)
        v8f acc[7];
#pragma unroll
        for (int c = 0; c < 7; ++c) { v8f z = {}; acc[c] = z; }

        const _Float16* aX = xh + ((size_t)mrow * L_ + t) * I_;
        const _Float16* aH = hcur + (size_t)mrow * D_;
        if (t + 1 < L_) __builtin_prefetch(xh + ((size_t)mrow * L_ + (t + 1)) * I_, 0, 0);

        for (int kk = 0; kk < K_; kk += 32) {
            const _Float16* ap = (kk < I_) ? (aX + kk) : (aH + (kk - I_));
            // A 16x32 f16: lane m holds K in {hlf*8..+7} U {16+hlf*8..+7}
            v8h alo = *(const v8h*)(ap + hlf * 8);
            v8h ahi = *(const v8h*)(ap + 16 + hlf * 8);
            v16h a;
#pragma unroll
            for (int q = 0; q < 8; ++q) { a[q] = alo[q]; a[8 + q] = ahi[q]; }
#pragma unroll
            for (int c = 0; c < 7; ++c) {
                // B 32x16 f16: lane n holds column n, K in {hlf*8..+7} U {16+hlf*8..+7}
                const _Float16* bp = smemh + (size_t)(c * 16 + n) * WPAD + kk;
                v8h blo = *(const v8h*)(bp + hlf * 8);
                v8h bhi = *(const v8h*)(bp + 16 + hlf * 8);
                v16h bv;
#pragma unroll
                for (int q = 0; q < 8; ++q) { bv[q] = blo[q]; bv[8 + q] = bhi[q]; }
                acc[c] = __builtin_amdgcn_wmma_f32_16x16x32_f16(
                    false, a, false, bv, (short)0, acc[c], false, false);
            }
        }

        // ---- Gates + state update + fused decay for h_{t+1}
        // C layout: row m = m0 + hlf*8 + r, col j = j0 + n  -> matches (bi[r], j).
#pragma unroll
        for (int r = 0; r < 8; ++r) {
            float i_g  = sigmoidf_(acc[0][r] + biasv[0]);
            float f_g  = sigmoidf_(acc[1][r] + biasv[1]);
            float ie_g = sigmoidf_(acc[2][r] + biasv[2]);
            float fe_g = sigmoidf_(acc[3][r] + biasv[3]);
            float o_n  = sigmoidf_(acc[4][r] + biasv[4]);
            float z    = tanhf(acc[5][r] + biasv[5]);
            float d_n  = softplusf_(acc[6][r] + biasv[6]);
            float c    = rc[r];                       // decayed cell for step t
            float cs_n = f_g * c + i_g * z;
            float ce_n = fe_g * rce[r] + ie_g * z;
            ro[r] = o_n; rcs[r] = cs_n; rce[r] = ce_n; rd[r] = d_n;
            if (t == la[r]) {                         // capture final state
                float* fo = outF + (size_t)bi[r] * 4 * D_ + (j0 + n);
                fo[0]      = o_n;
                fo[D_]     = cs_n;
                fo[2 * D_] = ce_n;
                fo[3 * D_] = d_n;
            }
            if (t + 1 < L_) {                         // uniform branch
                float dtn = ((t + 1) < sq[r]) ? dt[(size_t)bi[r] * L_ + t + 1] : 0.f;
                float cn  = ce_n + (cs_n - ce_n) * __expf(-d_n * dtn);
                rc[r] = cn;
                float hn = o_n * tanhf(cn);
                hnxt[ei[r]] = (_Float16)hn;
                out[((size_t)bi[r] * L_ + t + 1) * D_ + (j0 + n)] = hn;
            }
        }
        grid_sync(cnt, gen, NWG);
    }
}

extern "C" void kernel_launch(void* const* d_in, const int* in_sizes, int n_in,
                              void* d_out, int out_size, void* d_ws, size_t ws_size,
                              hipStream_t stream) {
    const float* x    = (const float*)d_in[0];
    const float* dt   = (const float*)d_in[1];
    const int*   seq  = (const int*)d_in[2];
    const float* bos  = (const float*)d_in[3];
    const float* W    = (const float*)d_in[4];
    const float* bias = (const float*)d_in[5];
    float* out = (float*)d_out;

    char* ws = (char*)d_ws;
    size_t off = 0;
    auto take = [&](size_t bytes) -> char* {
        char* p = ws + off;
        off += (bytes + 255) & ~(size_t)255;
        return p;
    };
    unsigned* bar = (unsigned*)take(256);
    _Float16* xh  = (_Float16*)take((size_t)B_ * L_ * I_ * 2);
    _Float16* wh  = (_Float16*)take((size_t)NC_ * D_ * K_ * 2);
    _Float16* hh0 = (_Float16*)take((size_t)B_ * D_ * 2);
    _Float16* hh1 = (_Float16*)take((size_t)B_ * D_ * 2);
    float* st_o   = (float*)take((size_t)B_ * D_ * 4);
    float* st_cs  = (float*)take((size_t)B_ * D_ * 4);
    float* st_ce  = (float*)take((size_t)B_ * D_ * 4);
    float* st_d   = (float*)take((size_t)B_ * D_ * 4);

    (void)hipMemsetAsync(bar, 0, 256, stream);
    prep_x<<<8192, 256, 0, stream>>>(x, seq, xh);
    prep_w<<<2688, 256, 0, stream>>>(W, wh);
    init_state<<<2, 256, 0, stream>>>(bos, W, bias, st_o, st_cs, st_ce, st_d);

    size_t smem = (size_t)NC_ * 16 * WPAD * 2;   // 173,824 B < 320 KB LDS
    (void)hipFuncSetAttribute((const void*)ct_lstm_persistent,
                              hipFuncAttributeMaxDynamicSharedMemorySize, (int)smem);
    ct_lstm_persistent<<<NWG, TPB, smem, stream>>>(xh, wh, hh0, hh1, st_o, st_cs,
                                                   st_ce, st_d, dt, seq, bias, out, bar);
}
